// BertSelfAttention_22153441313354
// MI455X (gfx1250) — matrix operand order repaired and checked by reading, not compiled, not run on hardware
//
#include <hip/hip_runtime.h>
#include <hip/hip_bf16.h>

// ---------------------------------------------------------------------------
// BERT self-attention + prefix gating, MI455X (gfx1250, wave32, WMMA).
//
//   (1) cvt_f32_to_bf16_x4 : fp32 -> bf16 (RNE) for hidden / prompt / Wq,k,v
//   (2) qkv_gemm_wmma<FULL>: [M,768] x [768,768]^T + bias via
//                            v_wmma_f32_16x16x32_bf16. FULL=true (M=8192)
//                            stages 128x32 tiles with double-buffered
//                            GLOBAL_LOAD_ASYNC_TO_LDS_B128 (ASYNCcnt);
//                            FULL=false (prompt, M=64) keeps guarded VGPR
//                            staging for OOB zero-fill.
//   (3) flash_attn_gated   : online-softmax attention (S=1024) + separately
//                            softmaxed 64-key prefix branch; K/V^T chunks
//                            double-buffered via async-to-LDS.
//
// Workspace: ~54 MB carved from d_ws.
// ---------------------------------------------------------------------------

typedef __bf16 bf16_t;
typedef bf16_t v16bf __attribute__((ext_vector_type(16)));
typedef float  v8f   __attribute__((ext_vector_type(8)));
typedef unsigned short u16;
typedef unsigned int   u32;
typedef unsigned long long u64;

#define H_    12
#define DH_   64
#define D_    768
#define B_    8
#define S_    1024
#define AT_   64
#define SCALE_ 0.125f  // 1/sqrt(64)

union Frag {               // one 16x32 bf16 A-frag or 32x16 bf16 B-frag
  uint4 q[2];              // 32 bytes = 16 bf16 per lane
  v16bf v;
};

__device__ __forceinline__ u16 f2bf(float f) {   // fp32 -> bf16, RNE
  u32 u = __float_as_uint(f);
  u32 r = u + 0x7FFFu + ((u >> 16) & 1u);
  return (u16)(r >> 16);
}

__device__ __forceinline__ v8f wmma_bf16(v16bf a, v16bf b, v8f c) {
  return __builtin_amdgcn_wmma_f32_16x16x32_bf16(
      false, a, false, b, (short)0, c, false, false);
}

__device__ __forceinline__ v8f zero8() {
  v8f z = {0.f, 0.f, 0.f, 0.f, 0.f, 0.f, 0.f, 0.f};
  return z;
}

// 16B global -> LDS, no VGPR data movement, tracked by ASYNCcnt.
__device__ __forceinline__ void async_b128(const void* gptr, void* lptr) {
  const u32 lds = (u32)(uintptr_t)lptr;         // low 32b of shared ptr = LDS addr
  const u64 gla = (u64)(uintptr_t)gptr;
  asm volatile("global_load_async_to_lds_b128 %0, %1, off"
               :: "v"(lds), "v"(gla) : "memory");
}

// ---------------------------------------------------------------------------
// (1) fp32 -> bf16 conversion, 4 elements / thread
// ---------------------------------------------------------------------------
__global__ void cvt_f32_to_bf16_x4(const float4* __restrict__ src,
                                   ushort4* __restrict__ dst, int n4) {
  int i = blockIdx.x * blockDim.x + threadIdx.x;
  if (i >= n4) return;
  float4 f = src[i];
  ushort4 o;
  o.x = f2bf(f.x); o.y = f2bf(f.y); o.z = f2bf(f.z); o.w = f2bf(f.w);
  dst[i] = o;
}

// ---------------------------------------------------------------------------
// (2) QKV projection GEMM: Y = X @ W^T + b
//     Block: 256 thr = 8 waves; tile 128x128; K-step 32; each wave owns a
//     64x32 sub-tile = 4x2 WMMA accumulators. W rows are B-matrix columns
//     -> all fragments are contiguous 16B loads.
// ---------------------------------------------------------------------------
template <bool FULL>
__launch_bounds__(256)
__global__ void qkv_gemm_wmma(const u16* __restrict__ X, int M,
                              const u16* __restrict__ Wq,
                              const u16* __restrict__ Wk,
                              const u16* __restrict__ Wv,
                              const float* __restrict__ bq,
                              const float* __restrict__ bk,
                              const float* __restrict__ bv,
                              int modeOffset, int isPrompt,
                              u16* __restrict__ qOut,   // [B,H,S,64]
                              u16* __restrict__ kOut,   // [B,H,S,64]
                              u16* __restrict__ vTOut,  // [B,H,64,S]
                              u16* __restrict__ pkOut,  // [H,64,64]
                              u16* __restrict__ pvTOut) // [H,64,64]
{
  const int mode = blockIdx.z + modeOffset;  // 0=Q 1=K 2=V
  const u16*   W    = (mode == 0) ? Wq : (mode == 1) ? Wk : Wv;
  const float* bias = (mode == 0) ? bq : (mode == 1) ? bk : bv;

  __shared__ __align__(16) u16 Xs[2][128][32];
  __shared__ __align__(16) u16 Ws[2][128][32];

  const int tid  = threadIdx.x;
  const int lane = tid & 31;
  const int wave = tid >> 5;
  const int wm   = wave >> 2;     // 0..1 : 64-row band
  const int wn   = wave & 3;      // 0..3 : 32-col band
  const int ln   = lane & 15;
  const int kh   = lane >> 4;     // K-half select per ISA fragment layout
  const int m0   = blockIdx.x * 128;
  const int n0   = blockIdx.y * 128;

  v8f acc[4][2];
#pragma unroll
  for (int i = 0; i < 4; ++i)
#pragma unroll
    for (int j = 0; j < 2; ++j) acc[i][j] = zero8();

  const int lr = tid >> 1;          // tile row this thread stages
  const int lc = (tid & 1) * 16;    // 16-element column half

  // async-stage one 128x32 X tile + 128x32 W tile into buffer `buf`
  auto issue = [&](int buf, int k0) {
    const u16* gx = X + (size_t)(m0 + lr) * D_ + k0 + lc;
    const u16* gw = W + (size_t)(n0 + lr) * D_ + k0 + lc;
    async_b128(gx,     &Xs[buf][lr][lc]);
    async_b128(gx + 8, &Xs[buf][lr][lc + 8]);
    async_b128(gw,     &Ws[buf][lr][lc]);
    async_b128(gw + 8, &Ws[buf][lr][lc + 8]);
  };

  if (FULL) issue(0, 0);

  for (int k0 = 0; k0 < D_; k0 += 32) {
    int bsel;
    if (FULL) {
      const int cur = (k0 >> 5) & 1;
      bsel = cur;
      if (k0 + 32 < D_) {
        issue(cur ^ 1, k0 + 32);                  // prefetch next tile
        asm volatile("s_wait_asynccnt 0x4" ::: "memory");  // cur resident
      } else {
        asm volatile("s_wait_asynccnt 0x0" ::: "memory");
      }
      __syncthreads();
    } else {
      // guarded VGPR staging (prompt: rows >= M zero-filled), buffer 0
      bsel = 0;
      uint4 xa = make_uint4(0, 0, 0, 0), xb = make_uint4(0, 0, 0, 0);
      const int gr = m0 + lr;
      if (gr < M) {
        const uint4* s = (const uint4*)(X + (size_t)gr * D_ + k0 + lc);
        xa = s[0]; xb = s[1];
      }
      const uint4* wsrc = (const uint4*)(W + (size_t)(n0 + lr) * D_ + k0 + lc);
      const uint4 wa = wsrc[0], wb = wsrc[1];
      __syncthreads();
      *(uint4*)&Xs[0][lr][lc]     = xa;
      *(uint4*)&Xs[0][lr][lc + 8] = xb;
      *(uint4*)&Ws[0][lr][lc]     = wa;
      *(uint4*)&Ws[0][lr][lc + 8] = wb;
      __syncthreads();
    }

    Frag bfr[2];
#pragma unroll
    for (int nf = 0; nf < 2; ++nf) {
      const int n = wn * 32 + nf * 16 + ln;       // B: col = lane&15
      bfr[nf].q[0] = *(const uint4*)&Ws[bsel][n][kh * 8];
      bfr[nf].q[1] = *(const uint4*)&Ws[bsel][n][16 + kh * 8];
    }
#pragma unroll
    for (int mf = 0; mf < 4; ++mf) {
      Frag afr;                                    // A: row = lane&15
      const int r = wm * 64 + mf * 16 + ln;
      afr.q[0] = *(const uint4*)&Xs[bsel][r][kh * 8];
      afr.q[1] = *(const uint4*)&Xs[bsel][r][16 + kh * 8];
#pragma unroll
      for (int nf = 0; nf < 2; ++nf)
        acc[mf][nf] = wmma_bf16(afr.v, bfr[nf].v, acc[mf][nf]);
    }

    if (FULL) __syncthreads();   // all waves done with bsel before it is
                                 // overwritten by next iteration's issue()
  }

  // epilogue: +bias, fp32->bf16, scatter to attention-friendly layouts
#pragma unroll
  for (int mf = 0; mf < 4; ++mf) {
#pragma unroll
    for (int nf = 0; nf < 2; ++nf) {
      const int n  = n0 + wn * 32 + nf * 16 + ln;
      const float bb = bias[n];
      const int h = n >> 6;
      const int d = n & 63;
#pragma unroll
      for (int j = 0; j < 8; ++j) {              // D: row = vgpr + 8*(lane>=16)
        const int m = m0 + wm * 64 + mf * 16 + kh * 8 + j;
        if (m >= M) continue;
        const u16 o = f2bf(acc[mf][nf][j] + bb);
        if (isPrompt) {
          if (mode == 1) pkOut [(h * AT_ + m) * DH_ + d] = o;
          else           pvTOut[(h * DH_ + d) * AT_ + m] = o;
        } else {
          const int b  = m >> 10;
          const int s  = m & 1023;
          const int bh = b * H_ + h;
          if (mode == 0)      qOut [((size_t)bh * S_ + s) * DH_ + d] = o;
          else if (mode == 1) kOut [((size_t)bh * S_ + s) * DH_ + d] = o;
          else                vTOut[((size_t)bh * DH_ + d) * S_ + s] = o;
        }
      }
    }
  }
}

// ---------------------------------------------------------------------------
// (3) flash attention + gated prefix branch.
//     Block = 256 thr = 8 waves; one (b,h), 128 query rows (16 per wave).
//     K [32x64] and V^T [64x32] chunks double-buffered via async-to-LDS.
// ---------------------------------------------------------------------------
__device__ __forceinline__ void issue_kv(const u16* __restrict__ kBase,
                                         const u16* __restrict__ vBase,
                                         int vStride, int t0,
                                         u16 (*Kt)[64], u16 (*Vt)[32],
                                         int tid) {
  const int r = tid >> 3, c = (tid & 7) * 8;      // K: 32x64 bf16, 16B/thr
  async_b128(kBase + (size_t)(t0 + r) * DH_ + c, &Kt[r][c]);
  const int r2 = tid >> 2, c2 = (tid & 3) * 8;    // V^T: 64x32 bf16, 16B/thr
  async_b128(vBase + (size_t)r2 * vStride + t0 + c2, &Vt[r2][c2]);
}

__device__ __forceinline__ void compute_chunk(
    int t0, const float* __restrict__ maskRow,    // nullptr -> no mask
    const Frag (&qa)[2],
    u16 (*Kt)[64], u16 (*Vt)[32], u16 (*Pw)[32],  // Pw = this wave's P tile
    float (&M8)[8], float (&L8)[8], v8f (&A4)[4],
    int ln, int kh)
{
  // scores: two 16-key tiles, K-dim = 64 (2 WMMA each)
  v8f sc[2];
#pragma unroll
  for (int tb = 0; tb < 2; ++tb) {
    sc[tb] = zero8();
#pragma unroll
    for (int ks = 0; ks < 2; ++ks) {
      Frag kb;
      const int t = tb * 16 + ln;                  // B: col = key
      kb.q[0] = *(const uint4*)&Kt[t][ks * 32 + kh * 8];
      kb.q[1] = *(const uint4*)&Kt[t][ks * 32 + 16 + kh * 8];
      sc[tb] = wmma_bf16(qa[ks].v, kb.v, sc[tb]);
    }
    const float mv = maskRow ? maskRow[t0 + tb * 16 + ln] : 0.f;
#pragma unroll
    for (int j = 0; j < 8; ++j) sc[tb][j] = sc[tb][j] * SCALE_ + mv;
  }

  // online softmax: row = vgpr + 8*(lane>=16); reduce over 16-lane halves
  float cm[8];
#pragma unroll
  for (int j = 0; j < 8; ++j) cm[j] = fmaxf(sc[0][j], sc[1][j]);
#pragma unroll
  for (int x = 1; x <= 8; x <<= 1)
#pragma unroll
    for (int j = 0; j < 8; ++j) cm[j] = fmaxf(cm[j], __shfl_xor(cm[j], x, 32));

  float scl[8];
#pragma unroll
  for (int j = 0; j < 8; ++j) {
    const float nm = fmaxf(M8[j], cm[j]);
    scl[j] = __expf(M8[j] - nm);
    M8[j]  = nm;
  }

  float rs[8];
#pragma unroll
  for (int j = 0; j < 8; ++j) rs[j] = 0.f;
#pragma unroll
  for (int tb = 0; tb < 2; ++tb)
#pragma unroll
    for (int j = 0; j < 8; ++j) {
      const float e = __expf(sc[tb][j] - M8[j]);
      rs[j] += e;
      Pw[kh * 8 + j][tb * 16 + ln] = f2bf(e);     // D-layout -> LDS row-major
    }
#pragma unroll
  for (int x = 1; x <= 8; x <<= 1)
#pragma unroll
    for (int j = 0; j < 8; ++j) rs[j] += __shfl_xor(rs[j], x, 32);
#pragma unroll
  for (int j = 0; j < 8; ++j) L8[j] = L8[j] * scl[j] + rs[j];

#pragma unroll
  for (int db = 0; db < 4; ++db)
#pragma unroll
    for (int j = 0; j < 8; ++j) A4[db][j] *= scl[j];

  // cross-lane LDS transpose hazard: LDS in-order per wave; stop compiler
  // reordering and drain DS before the A-fragment reads.
  asm volatile("" ::: "memory");
  asm volatile("s_wait_dscnt 0x0" ::: "memory");

  Frag pa;                                        // P as 16x32 A-fragment
  pa.q[0] = *(const uint4*)&Pw[ln][kh * 8];
  pa.q[1] = *(const uint4*)&Pw[ln][16 + kh * 8];
#pragma unroll
  for (int db = 0; db < 4; ++db) {
    Frag vb;
    const int d = db * 16 + ln;                   // B: col = d
    vb.q[0] = *(const uint4*)&Vt[d][kh * 8];
    vb.q[1] = *(const uint4*)&Vt[d][16 + kh * 8];
    A4[db] = wmma_bf16(pa.v, vb.v, A4[db]);
  }
}

__launch_bounds__(256)
__global__ void flash_attn_gated(const u16* __restrict__ Q,
                                 const u16* __restrict__ K,
                                 const u16* __restrict__ VT,
                                 const u16* __restrict__ PK,
                                 const u16* __restrict__ PVT,
                                 const float* __restrict__ mask,  // [B,S]
                                 const float* __restrict__ gate,  // [H]
                                 float* __restrict__ out)         // [B,S,768]
{
  __shared__ __align__(16) u16 Kt[2][32][64];
  __shared__ __align__(16) u16 Vt[2][64][32];
  __shared__ __align__(16) u16 Pb[8][16][32];

  const int tid  = threadIdx.x;
  const int lane = tid & 31;
  const int wave = tid >> 5;
  const int ln   = lane & 15;
  const int kh   = lane >> 4;

  const int bh = blockIdx.x;           // 0..95
  const int b  = bh / H_;
  const int h  = bh % H_;
  const int s0 = blockIdx.y * 128 + wave * 16;

  // Q fragments for this wave's 16 rows (live the whole kernel)
  Frag qa[2];
#pragma unroll
  for (int ks = 0; ks < 2; ++ks) {
    const u16* qrow = Q + ((size_t)bh * S_ + s0 + ln) * DH_ + ks * 32;
    qa[ks].q[0] = *(const uint4*)(qrow + kh * 8);
    qa[ks].q[1] = *(const uint4*)(qrow + 16 + kh * 8);
  }

  float mr[8], lr[8], pmr[8], plr[8];
  v8f oacc[4], pacc[4];
#pragma unroll
  for (int j = 0; j < 8; ++j) { mr[j] = -1e30f; lr[j] = 0.f;
                                pmr[j] = -1e30f; plr[j] = 0.f; }
#pragma unroll
  for (int d = 0; d < 4; ++d) { oacc[d] = zero8(); pacc[d] = zero8(); }

  // one attention phase: nCh 32-key chunks, double-buffered async staging
  auto run_phase = [&](const u16* kB, const u16* vB, int vStride, int nCh,
                       const float* maskRow, float (&M8)[8], float (&L8)[8],
                       v8f (&A4)[4]) {
    issue_kv(kB, vB, vStride, 0, Kt[0], Vt[0], tid);
#pragma unroll 1
    for (int c = 0; c < nCh; ++c) {
      const int cur = c & 1;
      if (c + 1 < nCh) {
        issue_kv(kB, vB, vStride, (c + 1) * 32, Kt[cur ^ 1], Vt[cur ^ 1], tid);
        asm volatile("s_wait_asynccnt 0x2" ::: "memory");  // cur resident
      } else {
        asm volatile("s_wait_asynccnt 0x0" ::: "memory");
      }
      __syncthreads();                    // all waves' async data visible
      compute_chunk(c * 32, maskRow, qa, Kt[cur], Vt[cur], Pb[wave],
                    M8, L8, A4, ln, kh);
      __syncthreads();                    // done with cur before overwrite
    }
  };

  // ---- main attention: 1024 keys, online softmax, additive mask ----
  run_phase(K + (size_t)bh * S_ * DH_, VT + (size_t)bh * DH_ * S_, S_,
            S_ / 32, mask + (size_t)b * S_, mr, lr, oacc);

  // ---- prefix branch: 64 prompt keys, separate softmax, no mask ----
  run_phase(PK + (size_t)h * AT_ * DH_, PVT + (size_t)h * DH_ * AT_, AT_,
            AT_ / 32, (const float*)nullptr, pmr, plr, pacc);

  // ---- combine + store fp32 output in final [B,S,H*64] layout ----
  const float tg = tanhf(gate[h]);
#pragma unroll
  for (int db = 0; db < 4; ++db) {
    const int d = h * DH_ + db * 16 + ln;
#pragma unroll
    for (int j = 0; j < 8; ++j) {
      const int s = s0 + kh * 8 + j;
      const float v = oacc[db][j] / lr[j] + tg * (pacc[db][j] / plr[j]);
      out[((size_t)b * S_ + s) * D_ + d] = v;
    }
  }
}

// ---------------------------------------------------------------------------
// host launcher
// ---------------------------------------------------------------------------
extern "C" void kernel_launch(void* const* d_in, const int* in_sizes, int n_in,
                              void* d_out, int out_size, void* d_ws,
                              size_t ws_size, hipStream_t stream) {
  (void)in_sizes; (void)n_in; (void)out_size; (void)ws_size;

  const float* hs   = (const float*)d_in[0];  // [8,1024,768]
  const float* pt   = (const float*)d_in[1];  // [1,64,768]
  const float* gate = (const float*)d_in[2];  // [12,1,1]
  const float* mask = (const float*)d_in[3];  // [8,1,1,1024]
  const float* Wq   = (const float*)d_in[4];
  const float* bq   = (const float*)d_in[5];
  const float* Wk   = (const float*)d_in[6];
  const float* bk   = (const float*)d_in[7];
  const float* Wv   = (const float*)d_in[8];
  const float* bv   = (const float*)d_in[9];
  float* out = (float*)d_out;

  char* p = (char*)d_ws;
  auto carve = [&](size_t bytes) -> void* {
    void* r = (void*)p;
    p += (bytes + 255) & ~(size_t)255;
    return r;
  };
  u16* hbf  = (u16*)carve((size_t)B_ * S_ * D_ * 2);
  u16* pbf  = (u16*)carve((size_t)AT_ * D_ * 2);
  u16* wqb  = (u16*)carve((size_t)D_ * D_ * 2);
  u16* wkb  = (u16*)carve((size_t)D_ * D_ * 2);
  u16* wvb  = (u16*)carve((size_t)D_ * D_ * 2);
  u16* qbf  = (u16*)carve((size_t)B_ * H_ * S_ * DH_ * 2);   // [B,H,S,64]
  u16* kbf  = (u16*)carve((size_t)B_ * H_ * S_ * DH_ * 2);   // [B,H,S,64]
  u16* vtb  = (u16*)carve((size_t)B_ * H_ * DH_ * S_ * 2);   // [B,H,64,S]
  u16* pkb  = (u16*)carve((size_t)H_ * AT_ * DH_ * 2);       // [H,64,64]
  u16* pvtb = (u16*)carve((size_t)H_ * DH_ * AT_ * 2);       // [H,64,64]

  auto cvt = [&](const float* s, u16* d, int n) {
    const int n4 = n / 4;
    cvt_f32_to_bf16_x4<<<(n4 + 255) / 256, 256, 0, stream>>>(
        (const float4*)s, (ushort4*)d, n4);
  };
  cvt(hs, hbf, B_ * S_ * D_);
  cvt(pt, pbf, AT_ * D_);
  cvt(Wq, wqb, D_ * D_);
  cvt(Wk, wkb, D_ * D_);
  cvt(Wv, wvb, D_ * D_);

  // QKV projections (async-to-LDS, double-buffered): grid.z selects {Q,K,V}
  qkv_gemm_wmma<true><<<dim3(64, 6, 3), 256, 0, stream>>>(
      hbf, B_ * S_, wqb, wkb, wvb, bq, bk, bv,
      /*modeOffset=*/0, /*isPrompt=*/0,
      qbf, kbf, vtb, nullptr, nullptr);

  // Prompt K/V projection (M=64, guarded staging): grid.z -> modes {K,V}
  qkv_gemm_wmma<false><<<dim3(1, 6, 2), 256, 0, stream>>>(
      pbf, AT_, wqb, wkb, wvb, bq, bk, bv,
      /*modeOffset=*/1, /*isPrompt=*/1,
      nullptr, nullptr, nullptr, pkb, pvtb);

  // Fused flash attention + gated prefix branch
  flash_attn_gated<<<dim3(B_ * H_, S_ / 128), 256, 0, stream>>>(
      qbf, kbf, vtb, pkb, pvtb, mask, gate, out);
}
